// SequenceWeight_58548994179787
// MI455X (gfx1250) — matrix operand order repaired and checked by reading, not compiled, not run on hardware
//
#include <hip/hip_runtime.h>
#include <hip/hip_bf16.h>

#define NSEQ 512
#define LLEN 256
#define DDIM 256
#define HEADS 8
#define HPAD 16
#define DHEAD 32
#define EPSV 1e-5f

typedef __attribute__((ext_vector_type(16))) _Float16 v16h;
typedef __attribute__((ext_vector_type(8)))  _Float16 v8h;
typedef __attribute__((ext_vector_type(8)))  float    v8f;
typedef __attribute__((ext_vector_type(4)))  float    v4f;

__device__ __forceinline__ float waveSum(float v) {
#pragma unroll
  for (int o = 16; o > 0; o >>= 1) v += __shfl_xor(v, o, 32);
  return v;
}
__device__ __forceinline__ float waveMax(float v) {
#pragma unroll
  for (int o = 16; o > 0; o >>= 1) v = fmaxf(v, __shfl_xor(v, o, 32));
  return v;
}

// ---------------------------------------------------------------------------
// Kernel 1: q[l, o] = scale * (LN(msa[0,0,l,:]) @ Wq^T + bq)   (L x 256, f32)
// ---------------------------------------------------------------------------
__global__ __launch_bounds__(256) void k_query(
    const float* __restrict__ msa, const float* __restrict__ Wq,
    const float* __restrict__ bq, const float* __restrict__ gamma,
    const float* __restrict__ beta, float* __restrict__ qws) {
  __shared__ float xln[DDIM];
  __shared__ float red[8];
  const int l = blockIdx.x, tid = threadIdx.x;
  const int lane = tid & 31, wave = tid >> 5;

  float x = msa[l * DDIM + tid];  // n = 0 row of sequence l
  float s = waveSum(x);
  if (lane == 0) red[wave] = s;
  __syncthreads();
  s = (lane < 8) ? red[lane] : 0.f;
  s = waveSum(s);
  const float mean = s * (1.0f / DDIM);
  __syncthreads();
  const float d = x - mean;
  float s2 = waveSum(d * d);
  if (lane == 0) red[wave] = s2;
  __syncthreads();
  s2 = (lane < 8) ? red[lane] : 0.f;
  s2 = waveSum(s2);
  const float rq = rsqrtf(s2 * (1.0f / DDIM) + EPSV);
  xln[tid] = d * rq * gamma[tid] + beta[tid];
  __syncthreads();

  float acc = 0.f;
  const float* wrow = Wq + tid * DDIM;
#pragma unroll 4
  for (int c = 0; c < DDIM; ++c) acc += xln[c] * wrow[c];
  qws[l * DDIM + tid] = (acc + bq[tid]) * 0.1767766952966369f;  // 1/sqrt(32)
}

// ---------------------------------------------------------------------------
// Kernel 2: W_eff[l, hp, c] = sum_d q[l, hp*32+d] * Wk[hp*32+d, c]  (f16,
// hp padded 8->16 with zeros);  qbk[l, hp] = q[l, hp, :] . bk[hp, :]
// ---------------------------------------------------------------------------
__global__ __launch_bounds__(256) void k_weff(
    const float* __restrict__ qws, const float* __restrict__ Wk,
    const float* __restrict__ bk, _Float16* __restrict__ weff,
    float* __restrict__ qbk) {
  __shared__ float qrow[DDIM];
  const int l = blockIdx.x, tid = threadIdx.x;
  qrow[tid] = qws[l * DDIM + tid];
  __syncthreads();

  const int c = tid;
  for (int hp = 0; hp < HPAD; ++hp) {
    float acc = 0.f;
    if (hp < HEADS) {
      const float* wbase = Wk + hp * DHEAD * DDIM + c;
#pragma unroll 8
      for (int d2 = 0; d2 < DHEAD; ++d2)
        acc += qrow[hp * DHEAD + d2] * wbase[d2 * DDIM];
    }
    weff[(l * HPAD + hp) * DDIM + c] = (_Float16)acc;
  }
  if (tid < HPAD) {
    float sb = 0.f;
    if (tid < HEADS)
      for (int d2 = 0; d2 < DHEAD; ++d2)
        sb += qrow[tid * DHEAD + d2] * bk[tid * DHEAD + d2];
    qbk[l * HPAD + tid] = sb;
  }
}

// ---------------------------------------------------------------------------
// Kernel 3: per l: LN rows, WMMA logits (16n x 16h) += A(16x256)*B(256x16),
// fused softmax over N=512, write attn. grid = L, block = 128 (4 waves).
// ---------------------------------------------------------------------------
__global__ __launch_bounds__(128) void k_attn(
    const float* __restrict__ msa, const _Float16* __restrict__ weff,
    const float* __restrict__ qbk, const float* __restrict__ gamma,
    const float* __restrict__ beta, float* __restrict__ out) {
  __shared__ __align__(16) _Float16 aTile[4][16 * DDIM];  // 32 KB
  __shared__ float lg[NSEQ * HEADS];                      // 16 KB
  __shared__ float gam[DDIM], bet[DDIM];                  // 2 KB
  const int l = blockIdx.x, tid = threadIdx.x;
  const int lane = tid & 31, wave = tid >> 5;

  for (int i = tid; i < DDIM; i += 128) { gam[i] = gamma[i]; bet[i] = beta[i]; }
  __syncthreads();

  const int col = lane & 15;  // N column (head) for B/C, row M for A
  const int hi  = lane >> 4;  // K-half selector (contiguous k order, same for A and B)

  // B fragments resident in VGPRs: lanes0-15 hold K=k0*32+[0..15] (col=lane),
  // lanes16-31 hold K=k0*32+[16..31]  -> 16 contiguous halves per fragment.
  v16h bfrag[8];
  {
    const _Float16* wb = weff + (l * HPAD + col) * DDIM + hi * 16;
#pragma unroll
    for (int k0 = 0; k0 < 8; ++k0) bfrag[k0] = *(const v16h*)(wb + k0 * 32);
  }
  const float cinit = qbk[l * HPAD + col];  // logits bias q.bk, bcast over rows

  _Float16* aw = &aTile[wave][0];
  for (int t = wave; t < NSEQ / 16; t += 4) {
    // ---- LayerNorm 16 rows (n = t*16 .. t*16+15) into LDS as f16 ----
    for (int r = 0; r < 16; ++r) {
      const int n = t * 16 + r;
      const v4f* rp = (const v4f*)(msa + ((size_t)n * LLEN + l) * DDIM);
      // streamed exactly once -> non-temporal load, don't pollute L2/WGP$
      const v4f v0 = __builtin_nontemporal_load(rp + lane * 2);
      const v4f v1 = __builtin_nontemporal_load(rp + lane * 2 + 1);
      // prefetch this wave's next tile (+64 sequences) to hide HBM latency
      if (n + 64 < NSEQ) {
        const float* pf = msa + ((size_t)(n + 64) * LLEN + l) * DDIM + lane * 8;
        __builtin_prefetch(pf, 0, 0);
      }
      float s = v0[0] + v0[1] + v0[2] + v0[3] + v1[0] + v1[1] + v1[2] + v1[3];
      float s2 = v0[0] * v0[0] + v0[1] * v0[1] + v0[2] * v0[2] + v0[3] * v0[3] +
                 v1[0] * v1[0] + v1[1] * v1[1] + v1[2] * v1[2] + v1[3] * v1[3];
      s = waveSum(s);
      s2 = waveSum(s2);
      const float m = s * (1.0f / DDIM);
      const float rq = rsqrtf(s2 * (1.0f / DDIM) - m * m + EPSV);
      const int cb = lane * 8;
      v8h hv;
      hv[0] = (_Float16)((v0[0] - m) * rq * gam[cb + 0] + bet[cb + 0]);
      hv[1] = (_Float16)((v0[1] - m) * rq * gam[cb + 1] + bet[cb + 1]);
      hv[2] = (_Float16)((v0[2] - m) * rq * gam[cb + 2] + bet[cb + 2]);
      hv[3] = (_Float16)((v0[3] - m) * rq * gam[cb + 3] + bet[cb + 3]);
      hv[4] = (_Float16)((v1[0] - m) * rq * gam[cb + 4] + bet[cb + 4]);
      hv[5] = (_Float16)((v1[1] - m) * rq * gam[cb + 5] + bet[cb + 5]);
      hv[6] = (_Float16)((v1[2] - m) * rq * gam[cb + 6] + bet[cb + 6]);
      hv[7] = (_Float16)((v1[3] - m) * rq * gam[cb + 7] + bet[cb + 7]);
      *(v8h*)(aw + r * DDIM + cb) = hv;
    }

    // ---- preload ALL A fragments (16x ds_load_b128 issued back-to-back),
    //      then 8 back-to-back WMMAs: one dscnt wait instead of eight ----
    // A layout: lanes0-15 M=lane, halves = K{k0*32+0..15};
    //           lanes16-31 M=lane-16, halves = K{+16..31}.
    const _Float16* ar = aw + col * DDIM + hi * 16;
    v16h afrag[8];
#pragma unroll
    for (int k0 = 0; k0 < 8; ++k0) {
      v8h a0 = *(const v8h*)(ar + k0 * 32);
      v8h a1 = *(const v8h*)(ar + k0 * 32 + 8);
      afrag[k0] = __builtin_shufflevector(a0, a1, 0, 1, 2, 3, 4, 5, 6, 7, 8, 9,
                                          10, 11, 12, 13, 14, 15);
    }
    v8f c = {cinit, cinit, cinit, cinit, cinit, cinit, cinit, cinit};
#pragma unroll
    for (int k0 = 0; k0 < 8; ++k0) {
      c = __builtin_amdgcn_wmma_f32_16x16x32_f16(
          false, afrag[k0], false, bfrag[k0], (short)0, c, false, false);
    }

    // C layout: VGPR j -> row M=j (lanes0-15) / M=j+8 (lanes16-31), N=col.
    if (col < HEADS) {
#pragma unroll
      for (int j = 0; j < 8; ++j) {
        const int nloc = j + hi * 8;
        lg[(t * 16 + nloc) * HEADS + col] = c[j];
      }
    }
  }
  __syncthreads();

  // ---- fused softmax over the N axis, one wave per head ----
  for (int h = wave; h < HEADS; h += 4) {
    float mx = -3.4e38f;
    for (int n = lane; n < NSEQ; n += 32) mx = fmaxf(mx, lg[n * HEADS + h]);
    mx = waveMax(mx);
    float sum = 0.f;
    for (int n = lane; n < NSEQ; n += 32) sum += __expf(lg[n * HEADS + h] - mx);
    sum = waveSum(sum);
    const float inv = 1.0f / sum;
    for (int n = lane; n < NSEQ; n += 32)
      out[((size_t)n * LLEN + l) * HEADS + h] = __expf(lg[n * HEADS + h] - mx) * inv;
  }
}

// ---------------------------------------------------------------------------
extern "C" void kernel_launch(void* const* d_in, const int* in_sizes, int n_in,
                              void* d_out, int out_size, void* d_ws,
                              size_t ws_size, hipStream_t stream) {
  const float* msa = (const float*)d_in[0];
  const float* Wq  = (const float*)d_in[1];
  const float* bq  = (const float*)d_in[2];
  const float* Wk  = (const float*)d_in[3];
  const float* bk  = (const float*)d_in[4];
  const float* gam = (const float*)d_in[5];
  const float* bet = (const float*)d_in[6];
  // d_in[7] = stride (no math effect; ignored)
  float* out = (float*)d_out;

  char* ws = (char*)d_ws;
  float*    qws  = (float*)ws;                                  // 256 KB
  float*    qbk  = (float*)(ws + (size_t)LLEN * DDIM * 4);      // 16 KB
  _Float16* weff = (_Float16*)(ws + (size_t)LLEN * DDIM * 4 +
                               (size_t)LLEN * HPAD * 4);        // 2 MB

  k_query<<<LLEN, 256, 0, stream>>>(msa, Wq, bq, gam, bet, qws);
  k_weff <<<LLEN, 256, 0, stream>>>(qws, Wk, bk, weff, qbk);
  k_attn <<<LLEN, 128, 0, stream>>>(msa, weff, qbk, gam, bet, out);
}
